// TransformerEncoderLayer_59150289601127
// MI455X (gfx1250) — matrix operand order repaired and checked by reading, not compiled, not run on hardware
//
#include <hip/hip_runtime.h>

// ---------------------------------------------------------------------------
// Transformer encoder layer for MI455X (gfx1250), wave32 + WMMA bf16.
// B=8, S=1024, D=512, N=8, H=64, FF=2048.  fp32 in/out, bf16 matrix path
// with f32 WMMA accumulation (v_wmma_f32_16x16x32_bf16).
// A-tiles staged to LDS via CDNA5 async direct-to-LDS copies (ASYNCcnt).
// Attention uses unnormalized-exp accumulation (softmax is shift-invariant;
// logits here are O(10), far from fp32 exp overflow) -> no per-iteration
// cross-lane reductions, only one final 16-lane sum reduce.
// ---------------------------------------------------------------------------

#define BATCH   8
#define SEQ     1024
#define DMODEL  512
#define NHEADS  8
#define HDIM    64
#define FFDIM   2048
#define MROWS   (BATCH * SEQ)   // 8192
#define EPSLN   1e-6f

typedef __attribute__((ext_vector_type(16))) __bf16 v16bf;
typedef __attribute__((ext_vector_type(8)))  float  v8f;
typedef __attribute__((ext_vector_type(4)))  unsigned int u32x4;

union FragBF { v16bf v; u32x4 u[2]; };

static __device__ inline v8f v8f_zero() {
  v8f z = {0.f, 0.f, 0.f, 0.f, 0.f, 0.f, 0.f, 0.f};
  return z;
}

// CDNA5 async direct-to-LDS copy: per-lane 16B  global -> LDS, ASYNCcnt.
// lds_off: LDS byte address (low 32 bits of the generic shared pointer).
static __device__ inline void async_copy_b128(unsigned lds_off, const void* gptr) {
  unsigned long long ga = (unsigned long long)(uintptr_t)gptr;
  asm volatile("global_load_async_to_lds_b128 %0, %1, off"
               :: "v"(lds_off), "v"(ga) : "memory");
}
static __device__ inline void async_drain() {
  asm volatile("s_wait_asynccnt 0x0" ::: "memory");
}

// ---------------------------------------------------------------------------
// fp32 -> bf16 conversion (weights)
// ---------------------------------------------------------------------------
__global__ void cvt_kernel(const float* __restrict__ in, __bf16* __restrict__ out, int n) {
  int i = blockIdx.x * blockDim.x + threadIdx.x;
  if (i < n) out[i] = (__bf16)in[i];
}

// ---------------------------------------------------------------------------
// LayerNorm over rows of DMODEL, bf16 output.  One block (256 thr) per row.
// ---------------------------------------------------------------------------
__global__ __launch_bounds__(256)
void ln_kernel(const float* __restrict__ x, const float* __restrict__ g,
               const float* __restrict__ bta, __bf16* __restrict__ out) {
  __shared__ float red[16];
  int row = blockIdx.x;
  int tid = threadIdx.x;
  const float* xr = x + (size_t)row * DMODEL;
  float2 v = *(const float2*)(xr + tid * 2);

  float s = v.x + v.y;
  #pragma unroll
  for (int o = 16; o >= 1; o >>= 1) s += __shfl_xor(s, o, 32);
  if ((tid & 31) == 0) red[tid >> 5] = s;
  __syncthreads();
  float mean = 0.f;
  #pragma unroll
  for (int i = 0; i < 8; i++) mean += red[i];
  mean *= (1.f / DMODEL);

  float d0 = v.x - mean, d1 = v.y - mean;
  float s2 = d0 * d0 + d1 * d1;
  #pragma unroll
  for (int o = 16; o >= 1; o >>= 1) s2 += __shfl_xor(s2, o, 32);
  if ((tid & 31) == 0) red[8 + (tid >> 5)] = s2;
  __syncthreads();
  float var = 0.f;
  #pragma unroll
  for (int i = 0; i < 8; i++) var += red[8 + i];
  var *= (1.f / DMODEL);

  float inv = rsqrtf(var + EPSLN);
  int c0 = tid * 2;
  out[(size_t)row * DMODEL + c0]     = (__bf16)(g[c0]     * (d0 * inv) + bta[c0]);
  out[(size_t)row * DMODEL + c0 + 1] = (__bf16)(g[c0 + 1] * (d1 * inv) + bta[c0 + 1]);
}

// ---------------------------------------------------------------------------
// bf16 GEMM: C[M,N] = A[M,K] * B[K,N]  (both row-major bf16).
// Block tile 128x128, K-step 32, 8 waves (4 along M x 2 along N),
// each wave: 2x4 grid of 16x16 WMMA accumulators (32x64).
// A tile [128][32] staged via async direct-to-LDS (stride-40 pad);
// B tile stored transposed [n][k] so B fragments are contiguous b128s.
// Compile-time epilogue: val = acc*scale (+bias[n]) (+resid[m,n]) (ReLU)
//   -> optional f32 out, optional bf16 out (row-major or [b,n,h,s] for V).
// ---------------------------------------------------------------------------
#define LDSP 40   // padded LDS stride (bf16 elems) -> conflict-free b128 reads

template <bool HASBIAS, bool HASRES, bool RELU, bool OUTF, bool OUTB, bool VT>
__global__ __launch_bounds__(256)
void gemm_bf16_kernel(const __bf16* __restrict__ A, const __bf16* __restrict__ Bm,
                      int M, int N, int K,
                      const float* __restrict__ bias,
                      const float* __restrict__ resid,
                      float scale,
                      float* __restrict__ outF, __bf16* __restrict__ outB) {
  __shared__ __attribute__((aligned(16))) __bf16 lA[128 * LDSP];
  __shared__ __attribute__((aligned(16))) __bf16 lB[128 * LDSP];  // [n][k]

  int tid  = threadIdx.x;
  int wave = tid >> 5, lane = tid & 31;
  int lm = lane & 15, lh = lane >> 4;
  int n0 = blockIdx.x * 128, m0 = blockIdx.y * 128;
  int wm = (wave & 3) * 32;   // wave M offset within tile
  int wn = (wave >> 2) * 64;  // wave N offset within tile

  v8f acc[2][4];
  #pragma unroll
  for (int i = 0; i < 2; i++)
    #pragma unroll
    for (int j = 0; j < 4; j++) acc[i][j] = v8f_zero();

  // per-thread staging coordinates
  int ar = tid >> 1, aseg = tid & 1;              // A: 2 thr/row, 16 bf16 each
  int bk = tid >> 3, bseg = tid & 7;              // B: 8 thr/row, 16 bf16 each
  unsigned lA_off = (unsigned)(uintptr_t)(lA + ar * LDSP + aseg * 16);

  for (int kk = 0; kk < K; kk += 32) {
    // ---- A tile 128x32: async global->LDS, 2 x b128 per thread ----
    {
      const __bf16* gp = A + (size_t)(m0 + ar) * K + kk + aseg * 16;
      async_copy_b128(lA_off, gp);
      async_copy_b128(lA_off + 16, gp + 8);
    }
    // ---- B tile transposed: global [K][N] -> LDS [n][k] ----
    {
      const u32x4* src = (const u32x4*)(Bm + (size_t)(kk + bk) * N + n0 + bseg * 16);
      union { u32x4 u[2]; __bf16 e[16]; } tmp;
      tmp.u[0] = src[0]; tmp.u[1] = src[1];
      if (kk + 32 < K)
        __builtin_prefetch(Bm + (size_t)(kk + 32 + bk) * N + n0 + bseg * 16, 0, 3);
      #pragma unroll
      for (int e = 0; e < 16; e++) lB[(bseg * 16 + e) * LDSP + bk] = tmp.e[e];
    }
    async_drain();          // ASYNCcnt -> 0 (our A-tile writes landed in LDS)
    __syncthreads();

    // ---- per-wave fragments + 8 WMMAs ----
    FragBF af[2];
    #pragma unroll
    for (int sm_ = 0; sm_ < 2; sm_++) {
      int row = wm + sm_ * 16 + lm;
      af[sm_].u[0] = *(const u32x4*)(lA + row * LDSP + lh * 8);
      af[sm_].u[1] = *(const u32x4*)(lA + row * LDSP + 16 + lh * 8);
    }
    FragBF bfr[4];
    #pragma unroll
    for (int sn = 0; sn < 4; sn++) {
      int col = wn + sn * 16 + lm;
      bfr[sn].u[0] = *(const u32x4*)(lB + col * LDSP + lh * 8);
      bfr[sn].u[1] = *(const u32x4*)(lB + col * LDSP + 16 + lh * 8);
    }
    #pragma unroll
    for (int sm_ = 0; sm_ < 2; sm_++)
      #pragma unroll
      for (int sn = 0; sn < 4; sn++)
        acc[sm_][sn] = __builtin_amdgcn_wmma_f32_16x16x32_bf16(
            false, af[sm_].v, false, bfr[sn].v, (short)0, acc[sm_][sn], false, false);
    __syncthreads();
  }

  // ---- epilogue (all flags compile-time) ----
  #pragma unroll
  for (int sm_ = 0; sm_ < 2; sm_++) {
    #pragma unroll
    for (int sn = 0; sn < 4; sn++) {
      int ncol = n0 + wn + sn * 16 + lm;
      float bv = HASBIAS ? bias[ncol] : 0.f;
      #pragma unroll
      for (int v = 0; v < 8; v++) {
        int mrow = m0 + wm + sm_ * 16 + v + 8 * lh;
        float val = acc[sm_][sn][v] * scale + bv;
        if (HASRES) val += resid[(size_t)mrow * N + ncol];
        if (RELU)   val = fmaxf(val, 0.f);
        if (OUTF)   outF[(size_t)mrow * N + ncol] = val;
        if (OUTB) {
          if (VT) {  // [b, n, h, s] layout (for V)
            int bidx = mrow / SEQ, srow = mrow % SEQ;
            int nh = ncol / HDIM, h = ncol % HDIM;
            outB[(((size_t)bidx * NHEADS + nh) * HDIM + h) * SEQ + srow] = (__bf16)val;
          } else {
            outB[(size_t)mrow * N + ncol] = (__bf16)val;
          }
        }
      }
    }
  }
}

// ---------------------------------------------------------------------------
// Attention: one wave per (b, head, 16-row f-tile).
// q,k: bf16 [b,s,n,h]  (q pre-scaled by H^-0.5);  vT: bf16 [b,n,h,s].
// logits tile via WMMA (Q x K^T, K^T fragments are contiguous h-runs).
// Unnormalized exp accumulate:  O = sum_t exp(S)*V,  rowsum = sum_t exp(S);
// exact softmax (shift-invariant), normalized once at the end.  P restaged
// through per-wave LDS (C layout -> A layout) with an s_wait_dscnt drain.
// ---------------------------------------------------------------------------
#define LDSP2 40

__global__ __launch_bounds__(256)
void attn_kernel(const __bf16* __restrict__ q, const __bf16* __restrict__ k,
                 const __bf16* __restrict__ vT, const float* __restrict__ abias,
                 __bf16* __restrict__ out) {
  __shared__ __attribute__((aligned(16))) __bf16 lp[8 * 16 * LDSP2];

  int tid  = threadIdx.x;
  int wave = tid >> 5, lane = tid & 31;
  int lm = lane & 15, lh = lane >> 4;
  int wid = blockIdx.x * 8 + wave;        // 4096 waves total
  int b  = wid >> 9;                       // / (NHEADS * SEQ/16)
  int n  = (wid >> 6) & (NHEADS - 1);
  int f0 = (wid & 63) << 4;
  __bf16* P = lp + wave * 16 * LDSP2;

  // Q A-fragments (16x64 = two 16x32 frags), rows f0..f0+15
  FragBF aq[2];
  const __bf16* qr = q + ((size_t)(b * SEQ + f0 + lm) * NHEADS + n) * HDIM;
  #pragma unroll
  for (int c = 0; c < 2; c++) {
    aq[c].u[0] = *(const u32x4*)(qr + c * 32 + lh * 8);
    aq[c].u[1] = *(const u32x4*)(qr + c * 32 + 16 + lh * 8);
  }

  v8f o[4];
  #pragma unroll
  for (int j = 0; j < 4; j++) o[j] = v8f_zero();
  v8f psum = v8f_zero();    // per-lane partial row sums of exp

  const __bf16* vb = vT + ((size_t)(b * NHEADS + n) * HDIM) * SEQ;

  for (int tp = 0; tp < SEQ / 32; tp++) {
    // ---- two 16x16 logit tiles (t-pair) ----
    #pragma unroll
    for (int half = 0; half < 2; half++) {
      int t0 = tp * 32 + half * 16;
      const __bf16* kr = k + ((size_t)(b * SEQ + t0 + lm) * NHEADS + n) * HDIM;
      v8f sacc = v8f_zero();
      #pragma unroll
      for (int c = 0; c < 2; c++) {   // h chunks of 32
        FragBF bk;
        bk.u[0] = *(const u32x4*)(kr + c * 32 + lh * 8);
        bk.u[1] = *(const u32x4*)(kr + c * 32 + 16 + lh * 8);
        sacc = __builtin_amdgcn_wmma_f32_16x16x32_bf16(
            false, aq[c].v, false, bk.v, (short)0, sacc, false, false);
      }
      float bt = abias[(size_t)b * SEQ + t0 + lm];
      // exp(logit + bias), accumulate row-sum, stash bf16 P into LDS
      #pragma unroll
      for (int v = 0; v < 8; v++) {
        float p = __expf(sacc[v] + bt);
        psum[v] += p;
        P[(v + 8 * lh) * LDSP2 + half * 16 + lm] = (__bf16)p;
      }
    }

    // same-wave LDS RAW: drain DScnt before re-reading as A fragment
    asm volatile("s_wait_dscnt 0x0" ::: "memory");

    FragBF pf;
    pf.u[0] = *(const u32x4*)(P + lm * LDSP2 + lh * 8);
    pf.u[1] = *(const u32x4*)(P + lm * LDSP2 + 16 + lh * 8);

    // ---- P(16x32) x V(32x64): 4 WMMAs, B frags contiguous in vT ----
    #pragma unroll
    for (int hs = 0; hs < 4; hs++) {
      const __bf16* vr = vb + (size_t)(hs * 16 + lm) * SEQ + tp * 32 + lh * 8;
      FragBF bv;
      bv.u[0] = *(const u32x4*)(vr);
      bv.u[1] = *(const u32x4*)(vr + 16);
      o[hs] = __builtin_amdgcn_wmma_f32_16x16x32_bf16(
          false, pf.v, false, bv.v, (short)0, o[hs], false, false);
    }
  }

  // ---- single final row-sum reduction (16 lanes per half), normalize, store
  float inv[8];
  #pragma unroll
  for (int v = 0; v < 8; v++) {
    float s = psum[v];
    s += __shfl_xor(s, 1, 32);
    s += __shfl_xor(s, 2, 32);
    s += __shfl_xor(s, 4, 32);
    s += __shfl_xor(s, 8, 32);
    inv[v] = 1.f / s;
  }
  #pragma unroll
  for (int hs = 0; hs < 4; hs++)
    #pragma unroll
    for (int v = 0; v < 8; v++) {
      int row = f0 + v + 8 * lh;
      int h = hs * 16 + lm;
      out[((size_t)(b * SEQ + row) * NHEADS + n) * HDIM + h] = (__bf16)(o[hs][v] * inv[v]);
    }
}

// ---------------------------------------------------------------------------
// Host-side launch
// ---------------------------------------------------------------------------
extern "C" void kernel_launch(void* const* d_in, const int* in_sizes, int n_in,
                              void* d_out, int out_size, void* d_ws, size_t ws_size,
                              hipStream_t stream) {
  const float* inputs = (const float*)d_in[0];
  const float* abias  = (const float*)d_in[1];
  const float* ln1_g  = (const float*)d_in[2];
  const float* ln1_b  = (const float*)d_in[3];
  const float* wq     = (const float*)d_in[4];
  const float* wk     = (const float*)d_in[5];
  const float* wv     = (const float*)d_in[6];
  const float* wo     = (const float*)d_in[7];
  const float* ln2_g  = (const float*)d_in[8];
  const float* ln2_b  = (const float*)d_in[9];
  const float* w1     = (const float*)d_in[10];
  const float* b1     = (const float*)d_in[11];
  const float* w2     = (const float*)d_in[12];
  const float* b2     = (const float*)d_in[13];
  float* out = (float*)d_out;

  char* ws = (char*)d_ws;
  size_t off = 0;
  auto take = [&](size_t bytes) -> char* {
    char* p = ws + off;
    off = (off + bytes + 255) & ~(size_t)255;
    return p;
  };

  __bf16* wq_b  = (__bf16*)take((size_t)DMODEL * DMODEL * 2);
  __bf16* wk_b  = (__bf16*)take((size_t)DMODEL * DMODEL * 2);
  __bf16* wv_b  = (__bf16*)take((size_t)DMODEL * DMODEL * 2);
  __bf16* wo_b  = (__bf16*)take((size_t)DMODEL * DMODEL * 2);
  __bf16* w1_b  = (__bf16*)take((size_t)DMODEL * FFDIM * 2);
  __bf16* w2_b  = (__bf16*)take((size_t)FFDIM * DMODEL * 2);
  __bf16* y1    = (__bf16*)take((size_t)MROWS * DMODEL * 2);
  __bf16* qb    = (__bf16*)take((size_t)MROWS * DMODEL * 2);
  __bf16* kb    = (__bf16*)take((size_t)MROWS * DMODEL * 2);
  __bf16* vtb   = (__bf16*)take((size_t)MROWS * DMODEL * 2);
  __bf16* attnb = (__bf16*)take((size_t)MROWS * DMODEL * 2);
  float*  x     = (float*) take((size_t)MROWS * DMODEL * 4);
  __bf16* y2    = (__bf16*)take((size_t)MROWS * DMODEL * 2);
  __bf16* hb    = (__bf16*)take((size_t)MROWS * FFDIM * 2);

  // 1) weight conversions
  const int CB = 256;
  cvt_kernel<<<(DMODEL * DMODEL + CB - 1) / CB, CB, 0, stream>>>(wq, wq_b, DMODEL * DMODEL);
  cvt_kernel<<<(DMODEL * DMODEL + CB - 1) / CB, CB, 0, stream>>>(wk, wk_b, DMODEL * DMODEL);
  cvt_kernel<<<(DMODEL * DMODEL + CB - 1) / CB, CB, 0, stream>>>(wv, wv_b, DMODEL * DMODEL);
  cvt_kernel<<<(DMODEL * DMODEL + CB - 1) / CB, CB, 0, stream>>>(wo, wo_b, DMODEL * DMODEL);
  cvt_kernel<<<(DMODEL * FFDIM + CB - 1) / CB, CB, 0, stream>>>(w1, w1_b, DMODEL * FFDIM);
  cvt_kernel<<<(FFDIM * DMODEL + CB - 1) / CB, CB, 0, stream>>>(w2, w2_b, FFDIM * DMODEL);

  // 2) LN1
  ln_kernel<<<MROWS, 256, 0, stream>>>(inputs, ln1_g, ln1_b, y1);

  // 3) QKV projections (q scaled by H^-0.5; v written transposed [b,n,h,s])
  dim3 gQKV(DMODEL / 128, MROWS / 128);
  gemm_bf16_kernel<false,false,false,false,true,false><<<gQKV, 256, 0, stream>>>(
      y1, wq_b, MROWS, DMODEL, DMODEL, nullptr, nullptr, 0.125f, nullptr, qb);
  gemm_bf16_kernel<false,false,false,false,true,false><<<gQKV, 256, 0, stream>>>(
      y1, wk_b, MROWS, DMODEL, DMODEL, nullptr, nullptr, 1.0f, nullptr, kb);
  gemm_bf16_kernel<false,false,false,false,true,true><<<gQKV, 256, 0, stream>>>(
      y1, wv_b, MROWS, DMODEL, DMODEL, nullptr, nullptr, 1.0f, nullptr, vtb);

  // 4) attention: 4096 waves = 512 blocks x 8 waves
  attn_kernel<<<BATCH * NHEADS * (SEQ / 16) / 8, 256, 0, stream>>>(qb, kb, vtb, abias, attnb);

  // 5) output projection + residual -> x (f32)
  gemm_bf16_kernel<false,true,false,true,false,false><<<gQKV, 256, 0, stream>>>(
      attnb, wo_b, MROWS, DMODEL, DMODEL, nullptr, inputs, 1.0f, x, nullptr);

  // 6) LN2
  ln_kernel<<<MROWS, 256, 0, stream>>>(x, ln2_g, ln2_b, y2);

  // 7) FFN1: relu(y2*w1 + b1) -> bf16
  dim3 gF1(FFDIM / 128, MROWS / 128);
  gemm_bf16_kernel<true,false,true,false,true,false><<<gF1, 256, 0, stream>>>(
      y2, w1_b, MROWS, FFDIM, DMODEL, b1, nullptr, 1.0f, nullptr, hb);

  // 8) FFN2: h*w2 + b2 + x -> out (f32)
  dim3 gF2(DMODEL / 128, MROWS / 128);
  gemm_bf16_kernel<true,true,false,true,false,false><<<gF2, 256, 0, stream>>>(
      hb, w2_b, MROWS, DMODEL, FFDIM, b2, x, 1.0f, out, nullptr);

  (void)in_sizes; (void)n_in; (void)out_size; (void)ws_size;
}
